// VirtualNodeEncoder_64201171140702
// MI455X (gfx1250) — matrix operand order repaired and checked by reading, not compiled, not run on hardware
//
#include <hip/hip_runtime.h>
#include <hip/hip_bf16.h>
#include <stdint.h>

#define DIM    512
#define DIM2   1024
#define NGRAPH 128
#define NLAYER 5
#define MAXDEP 20
#define BNEPS  1e-5f

typedef __attribute__((ext_vector_type(16))) __bf16 v16bf;
typedef __attribute__((ext_vector_type(8)))  float  v8f;

union Frag { unsigned u[8]; v16bf v; };

__device__ __forceinline__ unsigned short f2bf(float f) {
  union { float f; unsigned u; } x; x.f = f;
  unsigned r = x.u + 0x7FFFu + ((x.u >> 16) & 1u);
  return (unsigned short)(r >> 16);
}

// ---------------------------------------------------------------------------
// elementwise / graph kernels (memory bound; 23.3 TB/s path)
// ---------------------------------------------------------------------------
__global__ void k_fill(float* p, float v, long n) {
  long i = (long)blockIdx.x * blockDim.x + threadIdx.x;
  long st = (long)gridDim.x * blockDim.x;
  for (; i < n; i += st) p[i] = v;
}

__global__ void k_embed(const int* __restrict__ x, const int* __restrict__ nd,
                        const float* __restrict__ te, const float* __restrict__ ae,
                        const float* __restrict__ de, float* __restrict__ h, long total) {
  long i = (long)blockIdx.x * blockDim.x + threadIdx.x;
  long st = (long)gridDim.x * blockDim.x;
  for (; i < total; i += st) {
    long n = i >> 9; int d = (int)(i & (DIM - 1));
    int t = x[n * 2 + 0], a = x[n * 2 + 1];
    int dep = nd[n]; if (dep > MAXDEP) dep = MAXDEP;
    h[i] = te[(long)t * DIM + d] + ae[(long)a * DIM + d] + de[(long)dep * DIM + d];
  }
}

__global__ void k_deg(const int* __restrict__ ei, float* deg, int E) {
  int i = blockIdx.x * blockDim.x + threadIdx.x;
  if (i < E) atomicAdd(&deg[ei[i]], 1.0f);
}

__global__ void k_dis(const float* __restrict__ deg, float* dis, int N) {
  int i = blockIdx.x * blockDim.x + threadIdx.x;
  if (i < N) dis[i] = rsqrtf(deg[i]);
}

__global__ void k_initvn(const float* __restrict__ vnw, float* vn) {
  int i = blockIdx.x * blockDim.x + threadIdx.x;
  if (i < NGRAPH * DIM) vn[i] = vnw[i & (DIM - 1)];
}

__global__ void k_addvn(float* __restrict__ h, const float* __restrict__ vn,
                        const int* __restrict__ batch, long total) {
  long i = (long)blockIdx.x * blockDim.x + threadIdx.x;
  long st = (long)gridDim.x * blockDim.x;
  for (; i < total; i += st) {
    long n = i >> 9; int d = (int)(i & (DIM - 1));
    h[i] += vn[(long)batch[n] * DIM + d];
  }
}

__global__ void k_cvt(const float* __restrict__ h, unsigned short* __restrict__ hb, long total) {
  long i = (long)blockIdx.x * blockDim.x + threadIdx.x;
  long st = (long)gridDim.x * blockDim.x;
  for (; i < total; i += st) hb[i] = f2bf(h[i]);
}

__global__ void k_cvtwt(const float* __restrict__ w, unsigned short* __restrict__ wt) {
  int i = blockIdx.x * blockDim.x + threadIdx.x;   // i = n*512 + k
  if (i < DIM * DIM) {
    int n = i >> 9, k = i & (DIM - 1);
    wt[i] = f2bf(w[(long)k * DIM + n]);            // wt[n][k] = W[k][n]
  }
}

__global__ __launch_bounds__(256) void k_msg(
    const float* __restrict__ hx, const int* __restrict__ ei,
    const float* __restrict__ ea, const float* __restrict__ ew,
    const float* __restrict__ eb, const float* __restrict__ dis,
    float* __restrict__ agg, int E) {
  int e = blockIdx.x;
  if (e >= E) return;
  int r = ei[e], c = ei[E + e];
  float nrm = dis[r] * dis[c];
  float a0 = ea[(long)e * 2 + 0], a1 = ea[(long)e * 2 + 1];
  const float* hr = hx + (long)r * DIM;
  float* ac = agg + (long)c * DIM;
  for (int d = threadIdx.x; d < DIM; d += 256) {
    float v = hr[d] + a0 * ew[d] + a1 * ew[DIM + d] + eb[d];
    v = fmaxf(v, 0.f) * nrm;
    atomicAdd(&ac[d], v);
  }
}

__global__ void k_combine(float* __restrict__ h, const float* __restrict__ hx,
                          const float* __restrict__ cb, const float* __restrict__ deg,
                          long total) {
  long i = (long)blockIdx.x * blockDim.x + threadIdx.x;
  long st = (long)gridDim.x * blockDim.x;
  for (; i < total; i += st) {
    long n = i >> 9; int d = (int)(i & (DIM - 1));
    h[i] = h[i] + fmaxf(hx[i] + cb[d], 0.f) / deg[n];
  }
}

__global__ void k_colstats(const float* __restrict__ m, float* sum, float* ssq,
                           int rows, int cols) {
  int c = blockIdx.x * blockDim.x + threadIdx.x;
  if (c >= cols) return;
  float s = 0.f, s2 = 0.f;
  for (int r = blockIdx.y; r < rows; r += gridDim.y) {
    float v = m[(long)r * cols + c];
    s += v; s2 += v * v;
  }
  atomicAdd(&sum[c], s);
  atomicAdd(&ssq[c], s2);
}

__global__ void k_bn_apply(float* __restrict__ m, const float* __restrict__ sum,
                           const float* __restrict__ ssq, const float* __restrict__ g,
                           const float* __restrict__ b, long total, int colsMask,
                           float invR, int relu) {
  long i = (long)blockIdx.x * blockDim.x + threadIdx.x;
  long st = (long)gridDim.x * blockDim.x;
  for (; i < total; i += st) {
    int c = (int)(i & colsMask);   // cols is a power of two
    float mu = sum[c] * invR;
    float var = ssq[c] * invR - mu * mu;
    float v = (m[i] - mu) * rsqrtf(var + BNEPS) * g[c] + b[c];
    if (relu) v = fmaxf(v, 0.f);
    m[i] = v;
  }
}

__global__ void k_pool(const float* __restrict__ h, const int* __restrict__ batch,
                       float* __restrict__ vn, long total) {
  long i = (long)blockIdx.x * blockDim.x + threadIdx.x;
  long st = (long)gridDim.x * blockDim.x;
  for (; i < total; i += st) {
    long n = i >> 9; int d = (int)(i & (DIM - 1));
    atomicAdd(&vn[(long)batch[n] * DIM + d], h[i]);
  }
}

__global__ void k_dense(const float* __restrict__ in, const float* __restrict__ w,
                        const float* __restrict__ b, float* __restrict__ out,
                        int rows, int K, int cols) {
  long i = (long)blockIdx.x * blockDim.x + threadIdx.x;
  long total = (long)rows * cols;
  if (i >= total) return;
  int r = (int)(i / cols), c = (int)(i % cols);
  float s = b[c];
  for (int k = 0; k < K; ++k) s += in[(long)r * K + k] * w[(long)k * cols + c];
  out[i] = s;
}

// ---------------------------------------------------------------------------
// WMMA bf16 GEMM: out[N x 512] = bf16(h)[N x 512] @ Wt^T + bias  (f32 accum)
//   block 256 threads = 8 waves (2 M x 4 N), wave tile 32x64, block tile 64x256
//   K-step 32 (V_WMMA_F32_16X16X32_BF16), double-buffered async LDS tiles
// ---------------------------------------------------------------------------
#define TK 32
#define TM 64
#define TN 256
#define ASTR 40   // padded bf16 row stride in LDS (32 + 8)

__global__ __launch_bounds__(256) void k_gemm(
    const unsigned short* __restrict__ hb,   // N x 512, bf16 row-major
    const unsigned short* __restrict__ wt,   // 512 x 512, bf16, wt[n][k] = W[k][n]
    const float* __restrict__ bias,          // 512
    float* __restrict__ out,                 // N x 512, f32
    int Nrows) {
  __shared__ unsigned short As[2][TM][ASTR];
  __shared__ unsigned short Bs[2][TN][ASTR];

  const int tid  = threadIdx.x;
  const int lane = tid & 31;
  const int wave = tid >> 5;
  const int wm   = wave & 1;      // 0..1  -> M offset wm*32
  const int wn   = wave >> 1;     // 0..3  -> N offset wn*64
  const int m0   = blockIdx.y * TM;
  const int n0   = blockIdx.x * TN;
  const int rr   = lane & 15;
  const int hf   = lane >> 4;

  auto loadA = [&](int buf, int k0) {
    int row = tid >> 2, seg = tid & 3;                 // 64 rows x 4 x 16B
    long grow = m0 + row; if (grow >= Nrows) grow = Nrows - 1;
    unsigned long long ga = (unsigned long long)(hb + grow * DIM + k0 + seg * 8);
    unsigned lds = (unsigned)(size_t)(&As[buf][row][seg * 8]);
    asm volatile("global_load_async_to_lds_b128 %0, %1, off" :: "v"(lds), "v"(ga) : "memory");
  };
  auto loadB = [&](int buf, int k0) {
#pragma unroll
    for (int it = 0; it < 4; ++it) {                   // 256 rows x 4 x 16B
      int row = it * 64 + (tid >> 2), seg = tid & 3;
      unsigned long long ga = (unsigned long long)(wt + (long)(n0 + row) * DIM + k0 + seg * 8);
      unsigned lds = (unsigned)(size_t)(&Bs[buf][row][seg * 8]);
      asm volatile("global_load_async_to_lds_b128 %0, %1, off" :: "v"(lds), "v"(ga) : "memory");
    }
  };

  v8f acc[2][4];
#pragma unroll
  for (int mi = 0; mi < 2; ++mi)
#pragma unroll
    for (int ni = 0; ni < 4; ++ni)
#pragma unroll
      for (int v = 0; v < 8; ++v) acc[mi][ni][v] = 0.f;

  loadA(0, 0); loadB(0, 0);                            // 5 async ops / wave

  int buf = 0;
  const int KSTEPS = DIM / TK;                         // 16
  for (int kt = 0; kt < KSTEPS; ++kt) {
    if (kt) __syncthreads();                           // buf^1 free to rewrite
    bool pre = (kt + 1 < KSTEPS);
    if (pre) { loadA(buf ^ 1, (kt + 1) * TK); loadB(buf ^ 1, (kt + 1) * TK); }
    if (pre) { asm volatile("s_wait_asynccnt 5" ::: "memory"); }  // current tile done, next 5 in flight
    else     { asm volatile("s_wait_asynccnt 0" ::: "memory"); }
    __syncthreads();                                   // current buf visible to all waves

    Frag a[2], b4[4];
#pragma unroll
    for (int mi = 0; mi < 2; ++mi) {
      const unsigned* ap = (const unsigned*)&As[buf][wm * 32 + mi * 16 + rr][0];
#pragma unroll
      for (int v = 0; v < 8; ++v) {
        int k = ((v >= 4) ? 16 : 0) + hf * 8 + (v & 3) * 2;   // ISA 16-bit A 16x32 layout
        a[mi].u[v] = ap[k >> 1];
      }
    }
#pragma unroll
    for (int ni = 0; ni < 4; ++ni) {
      const unsigned* bp = (const unsigned*)&Bs[buf][wn * 64 + ni * 16 + rr][0];
#pragma unroll
      for (int v = 0; v < 8; ++v) {
        int k = ((v >= 4) ? 16 : 0) + hf * 8 + (v & 3) * 2;   // same k order as A (table order)
        b4[ni].u[v] = bp[k >> 1];
      }
    }
#pragma unroll
    for (int mi = 0; mi < 2; ++mi)
#pragma unroll
      for (int ni = 0; ni < 4; ++ni)
        acc[mi][ni] = __builtin_amdgcn_wmma_f32_16x16x32_bf16(
            false, a[mi].v, false, b4[ni].v, (short)0, acc[mi][ni], false, false);
    buf ^= 1;
  }

  // C/D layout: VGPR v -> row = hf*8 + v, col = lane&15
#pragma unroll
  for (int mi = 0; mi < 2; ++mi)
#pragma unroll
    for (int ni = 0; ni < 4; ++ni)
#pragma unroll
      for (int v = 0; v < 8; ++v) {
        int row = m0 + wm * 32 + mi * 16 + hf * 8 + v;
        int col = n0 + wn * 64 + ni * 16 + rr;
        if (row < Nrows) out[(long)row * DIM + col] = acc[mi][ni][v] + bias[col];
      }
}

// ---------------------------------------------------------------------------
extern "C" void kernel_launch(void* const* d_in, const int* in_sizes, int n_in,
                              void* d_out, int out_size, void* d_ws, size_t ws_size,
                              hipStream_t stream) {
  const int*   x          = (const int*)d_in[0];
  const int*   node_depth = (const int*)d_in[1];
  const int*   edge_index = (const int*)d_in[2];
  const int*   batch      = (const int*)d_in[3];
  const float* edge_attr  = (const float*)d_in[4];
  const float* type_emb   = (const float*)d_in[5];
  const float* attr_emb   = (const float*)d_in[6];
  const float* depth_emb  = (const float*)d_in[7];
  const float* vn_w       = (const float*)d_in[8];
  const float* lin_w      = (const float*)d_in[9];
  const float* lin_b      = (const float*)d_in[10];
  const float* conv_bias  = (const float*)d_in[11];
  const float* edge_w     = (const float*)d_in[12];
  const float* edge_b     = (const float*)d_in[13];
  const float* bn_g       = (const float*)d_in[14];
  const float* bn_b       = (const float*)d_in[15];
  const float* mlp_w1     = (const float*)d_in[16];
  const float* mlp_b1     = (const float*)d_in[17];
  const float* mlp_g      = (const float*)d_in[18];
  const float* mlp_bb     = (const float*)d_in[19];
  const float* mlp_w2     = (const float*)d_in[20];
  const float* mlp_b2     = (const float*)d_in[21];

  const int N = in_sizes[0] / 2;
  const int E = in_sizes[2] / 2;
  const long ND = (long)N * DIM;

  float* h = (float*)d_out;                 // final output doubles as h buffer
  char* wsp = (char*)d_ws;
  auto carve = [&](size_t bytes) { void* p = (void*)wsp; wsp += (bytes + 255) & ~(size_t)255; return p; };
  float*          hx  = (float*)carve(ND * 4);
  unsigned short* hb  = (unsigned short*)carve(ND * 2);
  unsigned short* wt  = (unsigned short*)carve((size_t)DIM * DIM * 2);
  float*          deg = (float*)carve((size_t)N * 4);
  float*          dis = (float*)carve((size_t)N * 4);
  float*          vn  = (float*)carve((size_t)NGRAPH * DIM * 4);
  float*          t   = (float*)carve((size_t)NGRAPH * DIM2 * 4);
  float*          sum = (float*)carve((size_t)DIM2 * 4);
  float*          ssq = (float*)carve((size_t)DIM2 * 4);

  const dim3 B(256);
  const int GB = 8192;

  k_embed<<<GB, B, 0, stream>>>(x, node_depth, type_emb, attr_emb, depth_emb, h, ND);
  k_fill<<<1024, B, 0, stream>>>(deg, 1.0f, N);
  k_deg<<<(E + 255) / 256, B, 0, stream>>>(edge_index, deg, E);
  k_dis<<<(N + 255) / 256, B, 0, stream>>>(deg, dis, N);
  k_initvn<<<(NGRAPH * DIM + 255) / 256, B, 0, stream>>>(vn_w, vn);

  const dim3 gemmGrid(DIM / TN, (N + TM - 1) / TM);

  for (int l = 0; l < NLAYER; ++l) {
    k_addvn<<<GB, B, 0, stream>>>(h, vn, batch, ND);
    k_cvt<<<GB, B, 0, stream>>>(h, hb, ND);
    k_cvtwt<<<(DIM * DIM + 255) / 256, B, 0, stream>>>(lin_w + (size_t)l * DIM * DIM, wt);
    k_gemm<<<gemmGrid, B, 0, stream>>>(hb, wt, lin_b + (size_t)l * DIM, hx, N);

    k_fill<<<GB, B, 0, stream>>>(h, 0.f, ND);          // h becomes the scatter accumulator
    k_msg<<<E, B, 0, stream>>>(hx, edge_index, edge_attr,
                               edge_w + (size_t)l * 2 * DIM, edge_b + (size_t)l * DIM,
                               dis, h, E);
    k_combine<<<GB, B, 0, stream>>>(h, hx, conv_bias + (size_t)l * DIM, deg, ND);

    k_fill<<<4, B, 0, stream>>>(sum, 0.f, DIM2);
    k_fill<<<4, B, 0, stream>>>(ssq, 0.f, DIM2);
    k_colstats<<<dim3((DIM + 255) / 256, 256), B, 0, stream>>>(h, sum, ssq, N, DIM);
    k_bn_apply<<<GB, B, 0, stream>>>(h, sum, ssq, bn_g + (size_t)l * DIM, bn_b + (size_t)l * DIM,
                                     ND, DIM - 1, 1.0f / N, (l < NLAYER - 1) ? 1 : 0);

    if (l < NLAYER - 1) {
      k_pool<<<GB, B, 0, stream>>>(h, batch, vn, ND);
      k_dense<<<(NGRAPH * DIM2 + 255) / 256, B, 0, stream>>>(
          vn, mlp_w1 + (size_t)l * DIM * DIM2, mlp_b1 + (size_t)l * DIM2, t, NGRAPH, DIM, DIM2);
      k_fill<<<4, B, 0, stream>>>(sum, 0.f, DIM2);
      k_fill<<<4, B, 0, stream>>>(ssq, 0.f, DIM2);
      k_colstats<<<dim3((DIM2 + 255) / 256, 128), B, 0, stream>>>(t, sum, ssq, NGRAPH, DIM2);
      k_bn_apply<<<(NGRAPH * DIM2 + 255) / 256, B, 0, stream>>>(
          t, sum, ssq, mlp_g + (size_t)l * DIM2, mlp_bb + (size_t)l * DIM2,
          (long)NGRAPH * DIM2, DIM2 - 1, 1.0f / NGRAPH, 1);
      k_dense<<<(NGRAPH * DIM + 255) / 256, B, 0, stream>>>(
          t, mlp_w2 + (size_t)l * DIM2 * DIM, mlp_b2 + (size_t)l * DIM, vn, NGRAPH, DIM2, DIM);
    }
  }
}
